// DeFormBlock_23922967838847
// MI455X (gfx1250) — compile-verified, matrix operand order checked by reading, not run on hardware
//
#include <hip/hip_runtime.h>

// ---------------- problem constants ----------------
#define BB   4
#define CC   64
#define HH   128
#define WW   128
#define OC   64
#define HW   (HH * WW)          // 16384
#define KKT  9                  // 3x3 taps
#define NOFF 18                 // 2*K*K offset channels

typedef __attribute__((ext_vector_type(16))) __bf16       v16bf;
typedef __attribute__((ext_vector_type(8)))  float        v8f;
typedef __attribute__((ext_vector_type(4)))  unsigned int v4u;
typedef __attribute__((ext_vector_type(8)))  int          v8i;
typedef __attribute__((ext_vector_type(4)))  int          v4i;

#define ROWS 72   // LDS row stride (bf16 elems): 144B rows -> 16B-aligned vector stores

// ---------------- helpers ----------------
__device__ __forceinline__ unsigned short f2bf(float f) {
  union { float f; unsigned int u; } v; v.f = f;
  unsigned int r = v.u + 0x7FFFu + ((v.u >> 16) & 1u);   // RNE
  return (unsigned short)(r >> 16);
}
__device__ __forceinline__ float bflo(unsigned int u) {
  union { unsigned int u; float f; } v; v.u = u << 16; return v.f;
}
__device__ __forceinline__ float bfhi(unsigned int u) {
  union { unsigned int u; float f; } v; v.u = u & 0xFFFF0000u; return v.f;
}
__device__ __forceinline__ unsigned int pack2(float lo, float hi) {
  return (unsigned int)f2bf(lo) | ((unsigned int)f2bf(hi) << 16);
}
__device__ __forceinline__ int iclamp127(int v) {
  return v < 0 ? 0 : (v > 127 ? 127 : v);
}

// CDNA5 16-bit 16x32 fragment layout: lane L holds row (L&15); K pairs for
// VGPR j are k = 8*(L>>4) + (j<4 ? 2j : 16+2(j-4)). Tile is [row][ROWS] bf16.
__device__ __forceinline__ v16bf load_frag(const unsigned short* tile, int rowBase, int lane) {
  const unsigned short* row = tile + (rowBase + (lane & 15)) * ROWS;
  int g8 = (lane >> 4) << 3;
  union { v16bf v; unsigned int u[8]; } fr;
#pragma unroll
  for (int j = 0; j < 8; ++j) {
    int kj = g8 + (j < 4 ? 2 * j : 16 + 2 * (j - 4));
    fr.u[j] = *(const unsigned int*)(row + kj);
  }
  return fr.v;
}

// ---------------- K0a: NCHW f32 -> NHWC bf16 transpose ----------------
__global__ __launch_bounds__(256) void transpose_x(const float* __restrict__ x,
                                                   unsigned short* __restrict__ x_t) {
  __shared__ unsigned short tile[64 * 65];
  const int tid = threadIdx.x;
  const int p0  = blockIdx.x * 64;
  const int b   = p0 >> 14;
  const int hw0 = p0 & (HW - 1);
  const float* xb = x + ((size_t)b << 20);
  for (int i = tid; i < 64 * 64; i += 256) {
    int c = i >> 6, j = i & 63;
    tile[j * 65 + c] = f2bf(xb[((size_t)c << 14) + hw0 + j]);
  }
  __syncthreads();
  for (int i = tid; i < 64 * 64; i += 256) {
    int j = i >> 6, c = i & 63;
    x_t[(((size_t)(p0 + j)) << 6) + c] = tile[j * 65 + c];
  }
}

// ---------------- K0b: pack weights to bf16, [tap][n][c] images --------------
__global__ __launch_bounds__(256) void pack_weights(const float* __restrict__ w_def,
                                                    const float* __restrict__ w_off,
                                                    unsigned short* __restrict__ w_def_bf,
                                                    unsigned short* __restrict__ w_off_bf) {
  const int tid = threadIdx.x;
  for (int i = tid; i < KKT * OC * CC; i += 256) {         // 36864
    int t = i >> 12, n = (i >> 6) & 63, c = i & 63;
    w_def_bf[i] = f2bf(w_def[((n << 6) | c) * KKT + t]);
  }
  for (int i = tid; i < KKT * 32 * CC; i += 256) {         // 18432
    int t = i >> 11, n = (i >> 6) & 31, c = i & 63;
    float v = (n < NOFF) ? w_off[(n * CC + c) * KKT + t] : 0.f;
    w_off_bf[i] = f2bf(v);
  }
}

// ---------------- K1: offset conv (WMMA + double-buffered async-LDS B) -------
__global__ __launch_bounds__(256) void off_conv(const unsigned short* __restrict__ x_t,
                                                const unsigned short* __restrict__ w_off_bf,
                                                const float* __restrict__ b_off,
                                                float* __restrict__ offset_out) {
  __shared__ alignas(16) unsigned short At[64 * ROWS];      // 64 pixels x 64 ch
  __shared__ alignas(16) unsigned short Bt[2 * 32 * ROWS];  // ping-pong 32x64 tiles
  const int tid  = threadIdx.x;
  const int lane = tid & 31;
  const int wave = tid >> 5;
  const int mblk = wave >> 1;                // 0..3 (M tile = 64)
  const int nblk = wave & 1;                 // 0..1 (N tile = 32)
  const int pix0 = blockIdx.x * 64;

  const unsigned ldsB   = (unsigned)(unsigned long long)&Bt[0];
  const unsigned ldsOff = ((tid >> 3) * (ROWS * 2)) + ((tid & 7) << 4);

  // async global->LDS: one b128 per thread covers one 32x128B tile per tap
  auto issueB = [&](int tt) {
    unsigned long long ga =
        (unsigned long long)(w_off_bf + (tt << 11)) + ((unsigned long long)tid << 4);
    unsigned ldst = ldsB + (unsigned)((tt & 1) * (32 * ROWS * 2)) + ldsOff;
    asm volatile("global_load_async_to_lds_b128 %0, %1, off"
                 :: "v"(ldst), "v"(ga) : "memory");
  };

  issueB(0);                                 // prefetch tap 0 under the A build

  v8f acc = {};
  for (int t = 0; t < KKT; ++t) {
    const int ky = t / 3 - 1, kx = t % 3 - 1;
    __syncthreads();                         // prior iteration's LDS reads retired
    // ---- A tile: one 3x3 tap, 16B (8 channels) per load, zero-padded border
#pragma unroll
    for (int rep = 0; rep < 2; ++rep) {
      int i = tid + rep * 256;
      int m = i >> 3, c0 = (i & 7) << 3;
      int p = pix0 + m;
      int b = p >> 14, h = (p >> 7) & 127, w = p & 127;
      int yy = h + ky, xx = w + kx;
      v4u q = {0u, 0u, 0u, 0u};
      if (yy >= 0 && yy < HH && xx >= 0 && xx < WW)
        q = *(const v4u*)(x_t + (((((size_t)b << 14) + (yy << 7) + xx) << 6) + c0));
      *(v4u*)(&At[m * ROWS + c0]) = q;
    }
    // ---- look-ahead DMA for next tap; in-order async => cnt<=1 means tap t landed
    if (t < KKT - 1) {
      issueB(t + 1);
      asm volatile("s_wait_asynccnt 0x1" ::: "memory");
    } else {
      asm volatile("s_wait_asynccnt 0x0" ::: "memory");
    }
    __syncthreads();
    const unsigned short* Bcur = Bt + (t & 1) * (32 * ROWS);
    acc = __builtin_amdgcn_wmma_f32_16x16x32_bf16(false, load_frag(At,        mblk * 16, lane),
                                                  false, load_frag(Bcur,      nblk * 16, lane),
                                                  (short)0, acc, false, false);
    acc = __builtin_amdgcn_wmma_f32_16x16x32_bf16(false, load_frag(At + 32,   mblk * 16, lane),
                                                  false, load_frag(Bcur + 32, nblk * 16, lane),
                                                  (short)0, acc, false, false);
  }
  const int n = nblk * 16 + (lane & 15);
  const int g = lane >> 4;
  if (n < NOFF) {
    float bo = b_off[n];
#pragma unroll
    for (int r = 0; r < 8; ++r) {
      int m = mblk * 16 + g * 8 + r;
      int p = pix0 + m;
      int b = p >> 14, hw = p & (HW - 1);
      offset_out[(((size_t)(b * NOFF + n)) << 14) + hw] = acc[r] + bo;
    }
  }
}

// ---------------- K2: deformable conv (bilinear im2col + WMMA + TDM B) -------
__global__ __launch_bounds__(256) void deform_gemm(const unsigned short* __restrict__ x_t,
                                                   const float* __restrict__ offset,
                                                   const unsigned short* __restrict__ w_def_bf,
                                                   const float* __restrict__ b_def,
                                                   float* __restrict__ y_raw) {
  __shared__ alignas(16) unsigned short At[32 * ROWS];      // 32 pixels x 64 ch
  __shared__ alignas(16) unsigned short Bt[2 * 64 * ROWS];  // ping-pong 64x64 tiles
  __shared__ float offs[32 * NOFF];
  const int tid  = threadIdx.x;
  const int lane = tid & 31;
  const int wave = tid >> 5;
  const int mblk = wave >> 2;                // 0..1 (M tile = 32)
  const int nblk = wave & 3;                 // 0..3 (N tile = 64)
  const int pix0 = blockIdx.x * 32;

  __builtin_prefetch(w_def_bf, 0, 0);        // global_prefetch_b8

#if __has_builtin(__builtin_amdgcn_tensor_load_to_lds)
  // TDM: 2D tensor, data_size=2B, 64x64 tile, stride0=64; pad_enable inserts
  // 4 dwords after every 32 dwords (one 128B row) -> LDS row stride 144B.
  auto issueTDM = [&](int tt) {
    unsigned long long ga = (unsigned long long)(w_def_bf + (tt << 12));
    unsigned ldsb = (unsigned)(unsigned long long)&Bt[(tt & 1) * (64 * ROWS)];
    v4u g0 = { 1u,                                         // count=1 valid descriptor
               ldsb,                                       // lds_addr
               (unsigned)ga,                               // global_addr[31:0]
               (unsigned)((ga >> 32) & 0x01FFFFFFu) | 0x80000000u };  // addr[56:32]|type=2
    v8i g1 = { (int)((1u << 16) | (1u << 20) | (4u << 22) | (3u << 25)),
               (int)(64u << 16),   // tensor_dim0 = 64
               (int)(64u << 16),   // tensor_dim1 = 64
               (int)(64u << 16),   // tile_dim0   = 64
               64,                 // tile_dim1   = 64, tile_dim2 = 0
               64,                 // tensor_dim0_stride = 64
               0, 0 };
    v4i z4 = {0, 0, 0, 0};
#if __clang_major__ >= 23
    v8i z8 = {0, 0, 0, 0, 0, 0, 0, 0};
    __builtin_amdgcn_tensor_load_to_lds(g0, g1, z4, z4, z8, 0);
#else
    __builtin_amdgcn_tensor_load_to_lds(g0, g1, z4, z4, 0);
#endif
  };
  if (wave == 0) issueTDM(0);                // prefetch tap 0 under offs staging
#endif

  for (int i = tid; i < 32 * NOFF; i += 256) {
    int m = i / NOFF, ch = i - m * NOFF;
    int p = pix0 + m;
    int b = p >> 14, hw = p & (HW - 1);
    offs[i] = offset[(((size_t)(b * NOFF + ch)) << 14) + hw];
  }

  const int am  = tid >> 3;                  // pixel 0..31
  const int ac0 = (tid & 7) << 3;            // channel group
  const int ap  = pix0 + am;
  const int ab  = ap >> 14, ah = (ap >> 7) & 127, aw = ap & 127;
  const unsigned short* xb = x_t + (((size_t)ab) << 20);

  v8f acc = {};
  for (int t = 0; t < KKT; ++t) {
    const int ky = t / 3, kx = t % 3;
    __syncthreads();   // offs ready (iter 0) / prior LDS reads retired (iter >0)

    // ---- A chunk: bilinear sample tap t, 8 channels per thread --------------
    {
      float py = (float)(ah - 1 + ky) + offs[am * NOFF + 2 * t];
      float px = (float)(aw - 1 + kx) + offs[am * NOFF + 2 * t + 1];
      float y0f = floorf(py), x0f = floorf(px);
      float wy1 = py - y0f, wy0 = 1.f - wy1;
      float wx1 = px - x0f, wx0 = 1.f - wx1;
      int y0 = (int)y0f, x0 = (int)x0f, y1 = y0 + 1, x1 = x0 + 1;
      float fy0 = (y0 >= 0 && y0 < HH) ? 1.f : 0.f;
      float fy1 = (y1 >= 0 && y1 < HH) ? 1.f : 0.f;
      float fx0 = (x0 >= 0 && x0 < WW) ? 1.f : 0.f;
      float fx1 = (x1 >= 0 && x1 < WW) ? 1.f : 0.f;
      float w00 = wy0 * wx0 * fy0 * fx0, w01 = wy0 * wx1 * fy0 * fx1;
      float w10 = wy1 * wx0 * fy1 * fx0, w11 = wy1 * wx1 * fy1 * fx1;
      int y0c = iclamp127(y0), y1c = iclamp127(y1);
      int x0c = iclamp127(x0), x1c = iclamp127(x1);
      v4u q00 = *(const v4u*)(xb + ((((y0c << 7) + x0c) << 6) + ac0));
      v4u q01 = *(const v4u*)(xb + ((((y0c << 7) + x1c) << 6) + ac0));
      v4u q10 = *(const v4u*)(xb + ((((y1c << 7) + x0c) << 6) + ac0));
      v4u q11 = *(const v4u*)(xb + ((((y1c << 7) + x1c) << 6) + ac0));
      v4u out;
#pragma unroll
      for (int j = 0; j < 4; ++j) {
        float slo = w00 * bflo(q00[j]) + w01 * bflo(q01[j]) +
                    w10 * bflo(q10[j]) + w11 * bflo(q11[j]);
        float shi = w00 * bfhi(q00[j]) + w01 * bfhi(q01[j]) +
                    w10 * bfhi(q10[j]) + w11 * bfhi(q11[j]);
        out[j] = pack2(slo, shi);
      }
      *(v4u*)(&At[am * ROWS + ac0]) = out;
    }

    // ---- look-ahead TDM for next tap; in-order TDM => cnt<=1 means tap t done
#if __has_builtin(__builtin_amdgcn_tensor_load_to_lds)
    if (wave == 0) {
      if (t < KKT - 1) {
        issueTDM(t + 1);
        __builtin_amdgcn_s_wait_tensorcnt((short)1);
      } else {
        __builtin_amdgcn_s_wait_tensorcnt((short)0);
      }
    }
#else
    for (int i = tid; i < 64 * 64; i += 256) {
      Bt[(t & 1) * (64 * ROWS) + (i >> 6) * ROWS + (i & 63)] = w_def_bf[(t << 12) + i];
    }
#endif
    __syncthreads();
    const unsigned short* Bcur = Bt + (t & 1) * (64 * ROWS);
    acc = __builtin_amdgcn_wmma_f32_16x16x32_bf16(false, load_frag(At,        mblk * 16, lane),
                                                  false, load_frag(Bcur,      nblk * 16, lane),
                                                  (short)0, acc, false, false);
    acc = __builtin_amdgcn_wmma_f32_16x16x32_bf16(false, load_frag(At + 32,   mblk * 16, lane),
                                                  false, load_frag(Bcur + 32, nblk * 16, lane),
                                                  (short)0, acc, false, false);
  }
  const int n = nblk * 16 + (lane & 15);
  const int g = lane >> 4;
  const float bd = b_def[n];
#pragma unroll
  for (int r = 0; r < 8; ++r) {
    int m = mblk * 16 + g * 8 + r;
    int p = pix0 + m;
    int b = p >> 14, hw = p & (HW - 1);
    y_raw[(((size_t)(b * OC + n)) << 14) + hw] = acc[r] + bd;
  }
}

// ---------------- K3: per-channel BN statistics (no atomics) ------------------
__global__ __launch_bounds__(256) void stats_kernel(const float* __restrict__ y_raw,
                                                    float* __restrict__ stats) {
  __shared__ float s_sum[256], s_sq[256];
  const int c = blockIdx.x, tid = threadIdx.x;
  float sum = 0.f, sq = 0.f;
  for (int i = tid; i < BB * HW; i += 256) {
    int b = i >> 14, hw = i & (HW - 1);
    float v = y_raw[(((size_t)(b * OC + c)) << 14) + hw];
    sum += v; sq += v * v;
  }
  s_sum[tid] = sum; s_sq[tid] = sq;
  __syncthreads();
  for (int off = 128; off > 0; off >>= 1) {
    if (tid < off) { s_sum[tid] += s_sum[tid + off]; s_sq[tid] += s_sq[tid + off]; }
    __syncthreads();
  }
  if (tid == 0) {
    const float invN = 1.f / (float)(BB * HW);
    float m  = s_sum[0] * invN;
    float vr = s_sq[0] * invN - m * m;
    stats[c]      = m;
    stats[64 + c] = rsqrtf(vr + 1e-5f);
  }
}

// ---------------- K4: BN affine + PReLU --------------------------------------
__global__ __launch_bounds__(256) void bn_prelu(const float* __restrict__ y_raw,
                                                const float* __restrict__ stats,
                                                const float* __restrict__ gamma,
                                                const float* __restrict__ beta,
                                                const float* __restrict__ prelu,
                                                float* __restrict__ out) {
  int idx = blockIdx.x * 256 + threadIdx.x;
  int c = (idx >> 14) & 63;
  float v = y_raw[idx];
  float yb = (v - stats[c]) * stats[64 + c] * gamma[c] + beta[c];
  out[idx] = yb >= 0.f ? yb : prelu[c] * yb;
}

// ---------------- host launcher ----------------------------------------------
extern "C" void kernel_launch(void* const* d_in, const int* in_sizes, int n_in,
                              void* d_out, int out_size, void* d_ws, size_t ws_size,
                              hipStream_t stream) {
  const float* x      = (const float*)d_in[0];
  const float* w_off  = (const float*)d_in[1];
  const float* b_off  = (const float*)d_in[2];
  const float* w_def  = (const float*)d_in[3];
  const float* b_def  = (const float*)d_in[4];
  const float* gamma  = (const float*)d_in[5];
  const float* beta   = (const float*)d_in[6];
  const float* prelu  = (const float*)d_in[7];
  float* out = (float*)d_out;

  // workspace layout (~30 MB):
  //   x_t      NHWC bf16 copy of x            8,388,608 B
  //   offs     offset conv result f32         4,718,592 B
  //   y_raw    pre-BN deform output f32      16,777,216 B
  //   stats    mean[64], rstd[64]                   512 B
  //   w_def_bf bf16 [t][n][c]                    73,728 B
  //   w_off_bf bf16 [t][n(32)][c]                36,864 B
  char* ws = (char*)d_ws;
  unsigned short* x_t      = (unsigned short*)ws;
  float*          offs     = (float*)(ws + 8388608);
  float*          y_raw    = (float*)(ws + 8388608 + 4718592);
  float*          stats    = (float*)(ws + 8388608 + 4718592 + 16777216);
  unsigned short* w_def_bf = (unsigned short*)(ws + 8388608 + 4718592 + 16777216 + 512);
  unsigned short* w_off_bf = (unsigned short*)(ws + 8388608 + 4718592 + 16777216 + 512 + 73728);

  transpose_x <<<(BB * HW) / 64, 256, 0, stream>>>(x, x_t);
  pack_weights<<<1, 256, 0, stream>>>(w_def, w_off, w_def_bf, w_off_bf);
  off_conv    <<<(BB * HW) / 64, 256, 0, stream>>>(x_t, w_off_bf, b_off, offs);
  deform_gemm <<<(BB * HW) / 32, 256, 0, stream>>>(x_t, offs, w_def_bf, b_def, y_raw);
  stats_kernel<<<OC, 256, 0, stream>>>(y_raw, stats);
  bn_prelu    <<<(BB * OC * HW) / 256, 256, 0, stream>>>(y_raw, stats, gamma, beta, prelu, out);
}
